// Attention_1812476199179
// MI455X (gfx1250) — matrix operand order repaired and checked by reading, not compiled, not run on hardware
//
#include <hip/hip_runtime.h>

// ---------------- problem constants ----------------
#define BATCH  4
#define SEQ    2048
#define DMODEL 1024
#define NHEAD  16
#define HDIM   64
#define D3     3072
#define ROWS   (BATCH * SEQ)   // 8192

#if __has_builtin(__builtin_amdgcn_global_load_async_to_lds_b128) && \
    __has_builtin(__builtin_amdgcn_s_wait_asynccnt)
#define ASYNC_COPY 1
#else
#define ASYNC_COPY 0
#endif

typedef __attribute__((ext_vector_type(16))) _Float16 v16h;
typedef __attribute__((ext_vector_type(8)))  _Float16 v8h;
typedef __attribute__((ext_vector_type(8)))  float    v8f;
typedef __attribute__((ext_vector_type(4)))  int      v4i;

union V16 { struct { v8h lo; v8h hi; } h; v16h v; };

#define NEG_INF (-1e30f)

// 16-byte global -> LDS copy; async (gfx1250 GLOBAL_LOAD_ASYNC_TO_LDS_B128,
// tracked by ASYNCcnt) when the builtin exists, else synchronous.
static __device__ __forceinline__ void cp16(_Float16* l, const _Float16* g) {
#if ASYNC_COPY
  __builtin_amdgcn_global_load_async_to_lds_b128(
      (__attribute__((address_space(1))) v4i*)(g),
      (__attribute__((address_space(3))) v4i*)(l), 0, 0);
#else
  *(uint4*)l = *(const uint4*)g;
#endif
}

template <int N> static __device__ __forceinline__ void wait_async() {
#if ASYNC_COPY
  __builtin_amdgcn_s_wait_asynccnt(N);
#endif
}

static __device__ __forceinline__ float rmax16(float v) {
  v = fmaxf(v, __shfl_xor(v, 1, 32));
  v = fmaxf(v, __shfl_xor(v, 2, 32));
  v = fmaxf(v, __shfl_xor(v, 4, 32));
  v = fmaxf(v, __shfl_xor(v, 8, 32));
  return v;
}
static __device__ __forceinline__ float rsum16(float v) {
  v += __shfl_xor(v, 1, 32);
  v += __shfl_xor(v, 2, 32);
  v += __shfl_xor(v, 4, 32);
  v += __shfl_xor(v, 8, 32);
  return v;
}

// ---------------- kernel 1a: x fp32 -> fp16 ----------------
__global__ void __launch_bounds__(256) cast_x_kernel(
    const float* __restrict__ x, _Float16* __restrict__ xh) {
  long i4 = ((long)blockIdx.x * blockDim.x + threadIdx.x) * 4;
  float4 f = *(const float4*)(x + i4);
  xh[i4 + 0] = (_Float16)f.x; xh[i4 + 1] = (_Float16)f.y;
  xh[i4 + 2] = (_Float16)f.z; xh[i4 + 3] = (_Float16)f.w;
}

// ---------------- kernel 1b: W fp32 -> fp16, transposed ----------------
// whT[col][k] = W[k][col]; 32x32 tiles through LDS, both sides coalesced.
__global__ void __launch_bounds__(256) transpose_w_kernel(
    const float* __restrict__ w, _Float16* __restrict__ whT) {
  __shared__ _Float16 t[32][33];
  const int kb = blockIdx.x * 32;   // DMODEL/32 = 32
  const int cb = blockIdx.y * 32;   // D3/32    = 96
  const int tx = threadIdx.x & 31, ty = threadIdx.x >> 5;
#pragma unroll
  for (int j = 0; j < 4; ++j) {
    int k = ty + j * 8;
    t[tx][k] = (_Float16)w[(size_t)(kb + k) * D3 + cb + tx];
  }
  __syncthreads();
#pragma unroll
  for (int j = 0; j < 4; ++j) {
    int c = ty + j * 8;
    whT[(size_t)(cb + c) * DMODEL + kb + tx] = t[c][tx];
  }
}

// ---------------- kernel 2: QKV projection GEMM ----------------
// C[8192,3072] = xh[8192,1024] @ whT[3072,1024]^T + bias.
// 128 threads = 4 waves, block tile 128x128, wave tile 64x64 (4x4 WMMA tiles)
// -> 16 ds_load_b128 per 16 WMMA (A and B fragments each reused 4x).
// Q -> qh[b,h,n,hd] (pre-scaled 1/8), K -> kh[b,h,n,hd], V -> vt[b,h,hd,n].
struct __align__(32) SmemGemm {
  _Float16 a[2][128 * 64];   // A tiles, [row][k]
  _Float16 b[2][128 * 64];   // B tiles, [col][k]
};                            // 64 KiB, double-buffered

static __device__ __forceinline__ void stage_gemm(
    SmemGemm& sm, int buf, const _Float16* __restrict__ xg,
    const _Float16* __restrict__ wg, int tid) {
#pragma unroll
  for (int it = 0; it < 8; ++it) {
    int c = tid + it * 128;              // 1024 8-half chunks per matrix
    int row = c >> 3, k8 = (c & 7) * 8;
    cp16(&sm.a[buf][row * 64 + k8], xg + (size_t)row * DMODEL + k8);
    cp16(&sm.b[buf][row * 64 + k8], wg + (size_t)row * DMODEL + k8);
  }
}

__global__ void __launch_bounds__(128) qkv_gemm_kernel(
    const _Float16* __restrict__ xh, const _Float16* __restrict__ whT,
    const float* __restrict__ bias,
    _Float16* __restrict__ qh, _Float16* __restrict__ kh,
    _Float16* __restrict__ vt) {
  __shared__ SmemGemm sm;
  const int tid  = threadIdx.x;
  const int w    = tid >> 5;           // 4 waves
  const int lane = tid & 31;
  const int ln16 = lane & 15;
  const int hi   = lane >> 4;
  const int rowbase = blockIdx.x * 128;
  const int colbase = blockIdx.y * 128;
  const int wr = (w & 1) * 64;         // 2x2 wave grid
  const int wc = (w >> 1) * 64;
  const _Float16* xg = xh + (size_t)rowbase * DMODEL;
  const _Float16* wg = whT + (size_t)colbase * DMODEL;

  const v8f vzero = {0.f, 0.f, 0.f, 0.f, 0.f, 0.f, 0.f, 0.f};
  v8f acc[4][4];
#pragma unroll
  for (int tm = 0; tm < 4; ++tm)
#pragma unroll
    for (int tn = 0; tn < 4; ++tn) acc[tm][tn] = vzero;

  const int NIT = DMODEL / 64;   // 16
  stage_gemm(sm, 0, xg, wg, tid);
  for (int i = 0; i < NIT; ++i) {
    const int cur = i & 1;
    if (i + 1 < NIT) {
      stage_gemm(sm, cur ^ 1, xg + (i + 1) * 64, wg + (i + 1) * 64, tid);
      wait_async<16>();          // buffer `cur` done; prefetch stays in flight
    } else {
      wait_async<0>();
    }
    __syncthreads();

    const _Float16* as = sm.a[cur];
    const _Float16* bs = sm.b[cur];
#pragma unroll
    for (int c = 0; c < 2; ++c) {        // two k32 sub-chunks
      V16 af[4];
#pragma unroll
      for (int tm = 0; tm < 4; ++tm) {
        int m = wr + tm * 16 + ln16;
        af[tm].h.lo = *(const v8h*)(as + m * 64 + c * 32 + hi * 8);
        af[tm].h.hi = *(const v8h*)(as + m * 64 + c * 32 + 16 + hi * 8);
      }
#pragma unroll
      for (int tn = 0; tn < 4; ++tn) {
        int n = wc + tn * 16 + ln16;
        V16 bfu;
        bfu.h.lo = *(const v8h*)(bs + n * 64 + c * 32 + hi * 8);
        bfu.h.hi = *(const v8h*)(bs + n * 64 + c * 32 + 16 + hi * 8);
        v16h bf = bfu.v;
#pragma unroll
        for (int tm = 0; tm < 4; ++tm)
          acc[tm][tn] = __builtin_amdgcn_wmma_f32_16x16x32_f16(
              false, af[tm].v, false, bf, (short)0, acc[tm][tn], false, false);
      }
    }
    __syncthreads();
  }

  // epilogue: bias, Q-scale, scatter (V transposed per head)
#pragma unroll
  for (int tn = 0; tn < 4; ++tn) {
    int gc   = colbase + wc + tn * 16 + ln16;
    int t    = gc >> 10;          // 0=Q 1=K 2=V
    int rem  = gc & 1023;
    int head = rem >> 6;
    int e    = rem & 63;
    float bv = bias[gc];
    float sc = (t == 0) ? 0.125f : 1.0f;   // fold 1/sqrt(64) into Q exactly
#pragma unroll
    for (int tm = 0; tm < 4; ++tm) {
      int grb = rowbase + wr + tm * 16 + hi * 8;
#pragma unroll
      for (int r = 0; r < 8; ++r) {
        int gr = grb + r;
        int bb = gr >> 11, nrow = gr & 2047;
        float val = (acc[tm][tn][r] + bv) * sc;
        size_t bh = (size_t)(bb * NHEAD + head);
        if (t == 2) {
          vt[(bh * HDIM + e) * SEQ + nrow] = (_Float16)val;
        } else {
          _Float16* dst = t ? kh : qh;
          dst[(bh * SEQ + nrow) * HDIM + e] = (_Float16)val;
        }
      }
    }
  }
}

// ---------------- kernel 3: causal flash attention ----------------
struct __align__(32) SmemAttn {
  _Float16 k[2][64 * 64];    // K blocks, [key][e]
  _Float16 vt[2][64 * 64];   // V blocks, [e][key] (already transposed)
  _Float16 p[8][16 * 64];    // per-wave P re-layout scratch
};                            // 48 KiB

static __device__ __forceinline__ void stage_attn(
    SmemAttn& sm, int buf, const _Float16* __restrict__ K,
    const _Float16* __restrict__ VT, int kbase, int tid) {
#pragma unroll
  for (int it = 0; it < 2; ++it) {
    int c = tid + it * 256;              // 512 8-half chunks per matrix
    int r = c >> 3, x8 = (c & 7) * 8;
    cp16(&sm.k[buf][r * 64 + x8], K + (size_t)(kbase + r) * HDIM + x8);
    cp16(&sm.vt[buf][r * 64 + x8], VT + (size_t)r * SEQ + kbase + x8);
  }
}

__global__ void __launch_bounds__(256) attn_kernel(
    const _Float16* __restrict__ qh, const _Float16* __restrict__ kh,
    const _Float16* __restrict__ vth, float* __restrict__ out) {
  __shared__ SmemAttn sm;
  const int tid  = threadIdx.x;
  const int w    = tid >> 5;
  const int lane = tid & 31;
  const int ln16 = lane & 15;
  const int hi   = lane >> 4;
  const int qb = blockIdx.x;        // 16 query blocks of 128
  const int bh = blockIdx.y;        // 64 (b,h) pairs
  const int b = bh >> 4, h = bh & 15;
  const _Float16* Q  = qh  + (size_t)bh * SEQ * HDIM;
  const _Float16* K  = kh  + (size_t)bh * SEQ * HDIM;
  const _Float16* VT = vth + (size_t)bh * HDIM * SEQ;
  const int qbase = qb * 128 + w * 16;

  // Q fragments stay in registers for the whole kernel
  V16 qf[2];
  {
    int m = qbase + ln16;
#pragma unroll
    for (int c = 0; c < 2; ++c) {
      qf[c].h.lo = *(const v8h*)(Q + (size_t)m * HDIM + c * 32 + hi * 8);
      qf[c].h.hi = *(const v8h*)(Q + (size_t)m * HDIM + c * 32 + 16 + hi * 8);
    }
  }

  const v8f vzero = {0.f, 0.f, 0.f, 0.f, 0.f, 0.f, 0.f, 0.f};
  float mst[8], lst[8];
  v8f o[4];
#pragma unroll
  for (int r = 0; r < 8; ++r) { mst[r] = NEG_INF; lst[r] = 0.f; }
#pragma unroll
  for (int te = 0; te < 4; ++te) o[te] = vzero;

  _Float16* pw = sm.p[w];
  const int nkb = (qb + 1) * 2;     // uniform per block; masking handles tail
  stage_attn(sm, 0, K, VT, 0, tid);
  for (int kb = 0; kb < nkb; ++kb) {
    const int cur = kb & 1;
    const int kbase = kb * 64;
    if (kb + 1 < nkb) {
      stage_attn(sm, cur ^ 1, K, VT, kbase + 64, tid);
      wait_async<4>();
    } else {
      wait_async<0>();
    }
    __syncthreads();
    const _Float16* ks = sm.k[cur];
    const _Float16* vs = sm.vt[cur];

    // S = Q @ K^T  (Q already carries the 1/8 scale)
    v8f s[4];
#pragma unroll
    for (int tn = 0; tn < 4; ++tn) s[tn] = vzero;
#pragma unroll
    for (int c = 0; c < 2; ++c) {
      v16h bf[4];
#pragma unroll
      for (int tn = 0; tn < 4; ++tn) {
        V16 t;
        t.h.lo = *(const v8h*)(ks + (tn * 16 + ln16) * 64 + c * 32 + hi * 8);
        t.h.hi = *(const v8h*)(ks + (tn * 16 + ln16) * 64 + c * 32 + 16 + hi * 8);
        bf[tn] = t.v;
      }
#pragma unroll
      for (int tn = 0; tn < 4; ++tn)
        s[tn] = __builtin_amdgcn_wmma_f32_16x16x32_f16(
            false, qf[c].v, false, bf[tn], (short)0, s[tn], false, false);
    }

    // causal mask (finite sentinel avoids inf-inf NaNs)
#pragma unroll
    for (int tn = 0; tn < 4; ++tn) {
      int kidx = kbase + tn * 16 + ln16;
#pragma unroll
      for (int r = 0; r < 8; ++r) {
        int row = qbase + r + hi * 8;
        if (kidx > row) s[tn][r] = NEG_INF;
      }
    }

    // online softmax: row m lives in VGPR m%8 across a 16-lane half-wave
    float alpha[8];
#pragma unroll
    for (int r = 0; r < 8; ++r) {
      float mx = fmaxf(fmaxf(s[0][r], s[1][r]), fmaxf(s[2][r], s[3][r]));
      mx = rmax16(mx);
      float mnew = fmaxf(mst[r], mx);
      float al = __expf(mst[r] - mnew);
      mst[r] = mnew;
      float rs = 0.f;
#pragma unroll
      for (int tn = 0; tn < 4; ++tn) {
        float pv = __expf(s[tn][r] - mnew);
        rs += pv;
        pw[(r + hi * 8) * 64 + tn * 16 + ln16] = (_Float16)pv;   // C->A relayout
      }
      rs = rsum16(rs);
      lst[r] = lst[r] * al + rs;
      alpha[r] = al;
    }
#pragma unroll
    for (int te = 0; te < 4; ++te)
#pragma unroll
      for (int r = 0; r < 8; ++r) o[te][r] *= alpha[r];

    // O += P @ V
#pragma unroll
    for (int c = 0; c < 2; ++c) {
      V16 pf;
      pf.h.lo = *(const v8h*)(pw + ln16 * 64 + c * 32 + hi * 8);
      pf.h.hi = *(const v8h*)(pw + ln16 * 64 + c * 32 + 16 + hi * 8);
#pragma unroll
      for (int te = 0; te < 4; ++te) {
        V16 vfu;
        vfu.h.lo = *(const v8h*)(vs + (te * 16 + ln16) * 64 + c * 32 + hi * 8);
        vfu.h.hi = *(const v8h*)(vs + (te * 16 + ln16) * 64 + c * 32 + 16 + hi * 8);
        v16h vf = vfu.v;
        o[te] = __builtin_amdgcn_wmma_f32_16x16x32_f16(
            false, pf.v, false, vf, (short)0, o[te], false, false);
      }
    }
    __syncthreads();
  }

  // normalize and write fp32 output in [b, n, h*hd] layout
#pragma unroll
  for (int te = 0; te < 4; ++te) {
    int e = h * HDIM + te * 16 + ln16;
#pragma unroll
    for (int r = 0; r < 8; ++r) {
      int row = qbase + r + hi * 8;
      out[((size_t)(b * SEQ + row)) * DMODEL + e] = o[te][r] / lst[r];
    }
  }
}

// ---------------- launch ----------------
extern "C" void kernel_launch(void* const* d_in, const int* in_sizes, int n_in,
                              void* d_out, int out_size, void* d_ws, size_t ws_size,
                              hipStream_t stream) {
  const float* x    = (const float*)d_in[0];
  const float* Wqkv = (const float*)d_in[1];
  const float* bias = (const float*)d_in[2];
  float* out = (float*)d_out;

  char* ws = (char*)d_ws;
  const size_t XH = (size_t)ROWS * DMODEL * sizeof(_Float16);               // 16 MiB
  const size_t WH = (size_t)DMODEL * D3 * sizeof(_Float16);                 //  6 MiB
  const size_t PH = (size_t)BATCH * NHEAD * SEQ * HDIM * sizeof(_Float16);  // 16 MiB
  _Float16* xh  = (_Float16*)(ws);
  _Float16* whT = (_Float16*)(ws + XH);
  _Float16* qh  = (_Float16*)(ws + XH + WH);
  _Float16* kh  = (_Float16*)(ws + XH + WH + PH);
  _Float16* vt  = (_Float16*)(ws + XH + WH + 2 * PH);

  cast_x_kernel<<<(int)((long)ROWS * DMODEL / 4 / 256), 256, 0, stream>>>(x, xh);
  transpose_w_kernel<<<dim3(DMODEL / 32, D3 / 32), 256, 0, stream>>>(Wqkv, whT);
  qkv_gemm_kernel<<<dim3(ROWS / 128, D3 / 128), 128, 0, stream>>>(
      xh, whT, bias, qh, kh, vt);
  attn_kernel<<<dim3(SEQ / 128, BATCH * NHEAD), 256, 0, stream>>>(
      qh, kh, vt, out);

  (void)in_sizes; (void)n_in; (void)out_size; (void)ws_size;
}
